// StationGAT_75230647157398
// MI455X (gfx1250) — matrix operand order repaired and checked by reading, not compiled, not run on hardware
//
#include <hip/hip_runtime.h>
#include <hip/hip_bf16.h>

typedef __attribute__((ext_vector_type(16))) _Float16 v16h;
typedef __attribute__((ext_vector_type(8)))  float    v8f;

// ---------------------------------------------------------------------------
// helpers
// ---------------------------------------------------------------------------
__device__ __forceinline__ float atomicMaxF(float* addr, float val) {
  // valid for arbitrary floats given init = -INF
  if (val >= 0.0f)
    return __int_as_float(atomicMax((int*)addr, __float_as_int(val)));
  else
    return __uint_as_float(atomicMin((unsigned int*)addr, __float_as_uint(val)));
}

__global__ void fill_f32(float* p, float v, long long n) {
  long long i = (long long)blockIdx.x * blockDim.x + threadIdx.x;
  if (i < n) p[i] = v;
}

// ---------------------------------------------------------------------------
// Fragment packing.
// A-fragment layout (16x32 f16, ISA 7.12.2): lane = 16*(K-half) + row,
//   element i -> K = ((i>>3)*16) + (((i>>1)&3)*2) + (i&1) + 8*(lane>>4)
// B-fragment layout (32x16 f16): lane = 16*(K-half) + col, element i -> K = ((i>>3)*16) + (i&7) + 8*(lane>>4)
// Packed storage: frag index f -> halfs [ (f*32 + lane)*16 .. +15 ]  (32B/lane)
// ---------------------------------------------------------------------------
__global__ void pack_a_k(const float* __restrict__ in, _Float16* __restrict__ out,
                         int M, int K) {
  int t = blockIdx.x * blockDim.x + threadIdx.x;   // (mt*KT + kt)*32 + lane
  const int KT = K >> 5;
  const int MT = (M + 15) >> 4;
  if (t >= MT * KT * 32) return;
  int lane = t & 31;
  int frag = t >> 5;
  int kt = frag % KT, mt = frag / KT;
  int row = mt * 16 + (lane & 15);
  if (row >= M) row = M - 1;                       // pad rows: dup (stores guarded)
  int khalf = (lane >> 4) * 8;
  const float* src = in + (size_t)row * K + kt * 32;
  v16h v;
#pragma unroll
  for (int p = 0; p < 8; ++p) {
    int kk = ((p & 4) << 2) + ((p & 3) << 1) + khalf;
    v[2 * p]     = (_Float16)src[kk];
    v[2 * p + 1] = (_Float16)src[kk + 1];
  }
  *(v16h*)(out + (size_t)t * 16) = v;
}

__global__ void pack_b_k(const float* __restrict__ in, _Float16* __restrict__ out,
                         int K, int N) {
  int t = blockIdx.x * blockDim.x + threadIdx.x;   // (kt*NT + nt)*32 + lane
  const int NT = N >> 4;
  const int KT = K >> 5;
  if (t >= KT * NT * 32) return;
  int lane = t & 31;
  int frag = t >> 5;
  int nt = frag % NT, kt = frag / NT;
  int col = nt * 16 + (lane & 15);
  int kbase = kt * 32 + (lane >> 4) * 8;
  v16h v;
#pragma unroll
  for (int i = 0; i < 16; ++i)
    v[i] = (_Float16)in[(size_t)(kbase + (i & 7) + ((i >> 3) << 4)) * N + col];
  *(v16h*)(out + (size_t)t * 16) = v;
}

// ---------------------------------------------------------------------------
// WMMA GEMM on packed fragments: C[M,N] f32 = A[M,K] f16 * B[K,N] f16.
// block = 128 threads (4 waves); wave handles 2 M-tiles x 4 N-tiles
// (8 accumulators, 8 WMMA per K-step vs 6 x 32B fragment loads).
// ---------------------------------------------------------------------------
__global__ void wmma_gemm_packed(const _Float16* __restrict__ Ap,
                                 const _Float16* __restrict__ Bp,
                                 float* __restrict__ C,
                                 int M, int N, int K) {
  const int wave = threadIdx.x >> 5;
  const int lane = threadIdx.x & 31;
  const int KT = K >> 5;
  const int NT = N >> 4;
  const int MT = (M + 15) >> 4;
  const int mt0 = (blockIdx.x * 4 + wave) * 2;
  const int nt0 = blockIdx.y * 4;
  if (mt0 >= MT) return;                           // wave-uniform: EXEC all ones
  const bool hasM1 = (mt0 + 1) < MT;

  const v16h* A0 = (const v16h*)Ap + (size_t)mt0 * KT * 32 + lane;
  const v16h* A1 = (const v16h*)Ap + (size_t)(hasM1 ? mt0 + 1 : mt0) * KT * 32 + lane;
  const v16h* Bb = (const v16h*)Bp + lane;

  v8f acc[2][4] = {};

  for (int kt = 0; kt < KT; ++kt) {
    v16h a0 = A0[(size_t)kt * 32];
    v16h a1 = A1[(size_t)kt * 32];
    size_t bf = ((size_t)kt * NT + nt0) * 32;
    v16h b0 = Bb[bf];
    v16h b1 = Bb[bf + 32];
    v16h b2 = Bb[bf + 64];
    v16h b3 = Bb[bf + 96];
    acc[0][0] = __builtin_amdgcn_wmma_f32_16x16x32_f16(false, a0, false, b0, (short)0, acc[0][0], false, false);
    acc[0][1] = __builtin_amdgcn_wmma_f32_16x16x32_f16(false, a0, false, b1, (short)0, acc[0][1], false, false);
    acc[0][2] = __builtin_amdgcn_wmma_f32_16x16x32_f16(false, a0, false, b2, (short)0, acc[0][2], false, false);
    acc[0][3] = __builtin_amdgcn_wmma_f32_16x16x32_f16(false, a0, false, b3, (short)0, acc[0][3], false, false);
    acc[1][0] = __builtin_amdgcn_wmma_f32_16x16x32_f16(false, a1, false, b0, (short)0, acc[1][0], false, false);
    acc[1][1] = __builtin_amdgcn_wmma_f32_16x16x32_f16(false, a1, false, b1, (short)0, acc[1][1], false, false);
    acc[1][2] = __builtin_amdgcn_wmma_f32_16x16x32_f16(false, a1, false, b2, (short)0, acc[1][2], false, false);
    acc[1][3] = __builtin_amdgcn_wmma_f32_16x16x32_f16(false, a1, false, b3, (short)0, acc[1][3], false, false);
  }

  // C layout (f32 16x16): VGPR r -> row r (lanes 0-15) / row r+8 (lanes 16-31)
  const int scol = nt0 * 16 + (lane & 15);
#pragma unroll
  for (int j = 0; j < 2; ++j) {
    if (j == 1 && !hasM1) break;
    const int srow = (mt0 + j) * 16 + (lane >> 4) * 8;
#pragma unroll
    for (int r = 0; r < 8; ++r) {
      int rr = srow + r;
      if (rr < M) {
        float* crow = C + (size_t)rr * N + scol;
        crow[0]  = acc[j][0][r];
        crow[16] = acc[j][1][r];
        crow[32] = acc[j][2][r];
        crow[48] = acc[j][3][r];
      }
    }
  }
}

// ---------------------------------------------------------------------------
// attention logits: one wave per (node, head); C == 128 (lane -> float4).
// ---------------------------------------------------------------------------
__global__ void alpha_k(const float* __restrict__ h,
                        const float* __restrict__ att_s,
                        const float* __restrict__ att_d,
                        float* __restrict__ al_s, float* __restrict__ al_d,
                        int N, int H) {
  const int C = 128;
  int wid  = (int)((blockIdx.x * (long long)blockDim.x + threadIdx.x) >> 5);
  int lane = threadIdx.x & 31;
  if (wid >= N * H) return;
  int n = wid / H, hd = wid % H;
  const float4* hp = (const float4*)(h + (size_t)n * H * C + (size_t)hd * C) + lane;
  const float4* as = (const float4*)(att_s + (size_t)hd * C) + lane;
  const float4* ad = (const float4*)(att_d + (size_t)hd * C) + lane;
  float4 v = *hp, a = *as, b = *ad;
  float s = v.x * a.x + v.y * a.y + v.z * a.z + v.w * a.w;
  float d = v.x * b.x + v.y * b.y + v.z * b.z + v.w * b.w;
#pragma unroll
  for (int o = 16; o > 0; o >>= 1) {
    s += __shfl_xor(s, o, 32);
    d += __shfl_xor(d, o, 32);
  }
  if (lane == 0) { al_s[wid] = s; al_d[wid] = d; }
}

// ---------------------------------------------------------------------------
// edge passes (E2 = E + N; edges >= E are self loops)
// ---------------------------------------------------------------------------
__global__ void edge_max_k(const int* __restrict__ ei, int E, int E2,
                           const float* __restrict__ al_s,
                           const float* __restrict__ al_d,
                           float* __restrict__ m, int H) {
  int e = blockIdx.x * blockDim.x + threadIdx.x;
  if (e >= E2) return;
  int s, d;
  if (e < E) { s = ei[e]; d = ei[E + e]; } else { s = d = e - E; }
  for (int hd = 0; hd < H; ++hd) {
    float v = al_s[(size_t)s * H + hd] + al_d[(size_t)d * H + hd];
    v = v > 0.f ? v : 0.2f * v;                    // leaky relu
    atomicMaxF(&m[(size_t)d * H + hd], v);
  }
}

__global__ void edge_expsum_k(const int* __restrict__ ei, int E, int E2,
                              const float* __restrict__ al_s,
                              const float* __restrict__ al_d,
                              const float* __restrict__ m,
                              float* __restrict__ ew, float* __restrict__ denom,
                              int H) {
  int e = blockIdx.x * blockDim.x + threadIdx.x;
  if (e >= E2) return;
  int s, d;
  if (e < E) { s = ei[e]; d = ei[E + e]; } else { s = d = e - E; }
  for (int hd = 0; hd < H; ++hd) {
    float v = al_s[(size_t)s * H + hd] + al_d[(size_t)d * H + hd];
    v = v > 0.f ? v : 0.2f * v;
    float ee = expf(v - m[(size_t)d * H + hd]);
    ew[(size_t)e * H + hd] = ee;
    atomicAdd(&denom[(size_t)d * H + hd], ee);
  }
}

// one block (128 threads) per edge, scatter weighted message into agg
__global__ void edge_agg_k(const int* __restrict__ ei, int E, int E2,
                           const float* __restrict__ h,
                           const float* __restrict__ ew,
                           const float* __restrict__ denom,
                           float* __restrict__ agg, int H, int C) {
  int e = blockIdx.x;
  if (e >= E2) return;
  int s, d;
  if (e < E) { s = ei[e]; d = ei[E + e]; } else { s = d = e - E; }
  const int HC = H * C;
  for (int c = threadIdx.x; c < HC; c += blockDim.x) {
    int hd = c / C;
    float w = ew[(size_t)e * H + hd] / (denom[(size_t)d * H + hd] + 1e-16f);
    atomicAdd(&agg[(size_t)d * HC + c], h[(size_t)s * HC + c] * w);
  }
}

// out = relu(agg + bias)  (f32 features; packing reads f32 directly)
__global__ void finalize_k(const float* __restrict__ agg,
                           const float* __restrict__ bias,
                           float* __restrict__ h32,
                           long long total, int HC) {
  long long i = (long long)blockIdx.x * blockDim.x + threadIdx.x;
  if (i >= total) return;
  int c = (int)(i % HC);
  float v = agg[i] + bias[c];
  h32[i] = v > 0.f ? v : 0.f;
}

// ---------------------------------------------------------------------------
// MLP head
// ---------------------------------------------------------------------------
__global__ void head1_k(const float* __restrict__ h, const float* __restrict__ W,
                        const float* __restrict__ b, float* __restrict__ z,
                        int N) {   // h [N,128], W [128,64], z [N,64]
  int idx = blockIdx.x * blockDim.x + threadIdx.x;
  if (idx >= N * 64) return;
  int n = idx / 64, j = idx % 64;
  const float* hp = h + (size_t)n * 128;
  float acc = b[j];
  for (int k = 0; k < 128; ++k) acc += hp[k] * W[k * 64 + j];
  z[idx] = acc > 0.f ? acc : 0.f;
}

__global__ void head2_k(const float* __restrict__ z, const float* __restrict__ W,
                        const float* __restrict__ b, float* __restrict__ out,
                        int N) {   // z [N,64], W [64,4], out [N,4]
  int idx = blockIdx.x * blockDim.x + threadIdx.x;
  if (idx >= N * 4) return;
  int n = idx / 4, j = idx % 4;
  const float* zp = z + (size_t)n * 64;
  float acc = b[j];
  for (int k = 0; k < 64; ++k) acc += zp[k] * W[k * 4 + j];
  out[idx] = acc;
}

// ---------------------------------------------------------------------------
// launch
// ---------------------------------------------------------------------------
extern "C" void kernel_launch(void* const* d_in, const int* in_sizes, int n_in,
                              void* d_out, int out_size, void* d_ws, size_t ws_size,
                              hipStream_t stream) {
  const float* x    = (const float*)d_in[0];
  const int*   ei   = (const int*)d_in[1];
  const float* W0   = (const float*)d_in[2];
  const float* as0  = (const float*)d_in[3];
  const float* ad0  = (const float*)d_in[4];
  const float* b0   = (const float*)d_in[5];
  const float* W1   = (const float*)d_in[6];
  const float* as1  = (const float*)d_in[7];
  const float* ad1  = (const float*)d_in[8];
  const float* b1   = (const float*)d_in[9];
  const float* W2   = (const float*)d_in[10];
  const float* as2  = (const float*)d_in[11];
  const float* ad2  = (const float*)d_in[12];
  const float* b2   = (const float*)d_in[13];
  const float* hW1  = (const float*)d_in[14];
  const float* hb1  = (const float*)d_in[15];
  const float* hW2  = (const float*)d_in[16];
  const float* hb2  = (const float*)d_in[17];
  float* out = (float*)d_out;

  const int N  = in_sizes[0] / 128;      // 50000
  const int E  = in_sizes[1] / 2;        // 800000
  const int E2 = E + N;                  // + self loops
  const int H = 4, HC = 512;
  const int MT = (N + 15) >> 4;          // M tiles

  // -------- workspace carve-out --------
  char* base = (char*)d_ws;
  size_t off = 0;
  auto carve = [&](size_t bytes) -> char* {
    char* p = base + off;
    off = (off + bytes + 255) & ~(size_t)255;
    return p;
  };
  _Float16* apack = (_Float16*)carve((size_t)MT * 16 * HC * 2);  // padded A frags
  _Float16* bpack = (_Float16*)carve((size_t)512 * 512 * 2);
  float* h     = (float*)carve((size_t)N * HC * 4);
  float* agg   = (float*)carve((size_t)N * HC * 4);
  float* al_s  = (float*)carve((size_t)N * H * 4);
  float* al_d  = (float*)carve((size_t)N * H * 4);
  float* m     = (float*)carve((size_t)N * H * 4);
  float* denom = (float*)carve((size_t)N * H * 4);
  float* ew    = (float*)carve((size_t)E2 * H * 4);
  float* z1    = (float*)carve((size_t)N * 64 * 4);
  (void)ws_size; (void)n_in; (void)out_size;

  const int TPB = 256;
  auto blk = [&](long long n) { return (unsigned)((n + TPB - 1) / TPB); };

  struct LayerDesc {
    const float *W, *as, *ad, *bias;
    int Kin, Nout, Hh;
  } layers[3] = {
    { W0, as0, ad0, b0, 128, 512, 4 },
    { W1, as1, ad1, b1, 512, 512, 4 },
    { W2, as2, ad2, b2, 512, 128, 1 },
  };

  for (int l = 0; l < 3; ++l) {
    const LayerDesc& L = layers[l];
    const int lhc = L.Hh * 128;
    const int KT = L.Kin >> 5, NT = L.Nout >> 4;

    // pack A (f32 features -> f16 fragments) and B (weights)
    const float* feats = (l == 0) ? x : h;
    pack_a_k<<<blk((long long)MT * KT * 32), TPB, 0, stream>>>(feats, apack, N, L.Kin);
    pack_b_k<<<blk((long long)KT * NT * 32), TPB, 0, stream>>>(L.W, bpack, L.Kin, L.Nout);

    // WMMA GEMM: h = feats @ W
    dim3 ggrid((MT + 7) / 8, L.Nout / 64);
    wmma_gemm_packed<<<ggrid, 128, 0, stream>>>(apack, bpack, h, N, L.Nout, L.Kin);

    // attention logits
    alpha_k<<<blk((long long)N * L.Hh * 32), TPB, 0, stream>>>(h, L.as, L.ad,
                                                               al_s, al_d, N, L.Hh);
    // segment state init
    fill_f32<<<blk((long long)N * L.Hh), TPB, 0, stream>>>(m, -INFINITY, (long long)N * L.Hh);
    fill_f32<<<blk((long long)N * L.Hh), TPB, 0, stream>>>(denom, 0.f, (long long)N * L.Hh);
    fill_f32<<<blk((long long)N * lhc), TPB, 0, stream>>>(agg, 0.f, (long long)N * lhc);

    // edge passes
    edge_max_k<<<blk(E2), TPB, 0, stream>>>(ei, E, E2, al_s, al_d, m, L.Hh);
    edge_expsum_k<<<blk(E2), TPB, 0, stream>>>(ei, E, E2, al_s, al_d, m, ew, denom, L.Hh);
    edge_agg_k<<<E2, 128, 0, stream>>>(ei, E, E2, h, ew, denom, agg, L.Hh, 128);

    // bias + relu -> f32 features for next layer / head
    finalize_k<<<blk((long long)N * lhc), TPB, 0, stream>>>(
        agg, L.bias, h, (long long)N * lhc, lhc);
  }

  // MLP head: h [N,128] -> z1 [N,64] -> out [N,4]
  head1_k<<<blk((long long)N * 64), TPB, 0, stream>>>(h, hW1, hb1, z1, N);
  head2_k<<<blk((long long)N * 4), TPB, 0, stream>>>(z1, hW2, hb2, out, N);
}
